// MultiheadSelfAttention_39608188404063
// MI455X (gfx1250) — matrix operand order repaired and checked by reading, not compiled, not run on hardware
//
#include <hip/hip_runtime.h>
#include <hip/hip_bf16.h>

// MI455X / gfx1250, wave32. All matrix math through v_wmma_f32_16x16x32_bf16.

typedef __attribute__((ext_vector_type(16))) __bf16 v16bf;
typedef __attribute__((ext_vector_type(8)))  __bf16 v8bf;
typedef __attribute__((ext_vector_type(8)))  float  v8f;

#define LSEQ 4096
#define CDIM 1024
#define HN   16
#define DH   64

__device__ __forceinline__ v16bf pack16(v8bf lo, v8bf hi) {
    v16bf r;
#pragma unroll
    for (int i = 0; i < 8; ++i) { r[i] = lo[i]; r[i + 8] = hi[i]; }
    return r;
}

__device__ __forceinline__ v8f wmma_bf16(v16bf a, v16bf b, v8f c) {
    return __builtin_amdgcn_wmma_f32_16x16x32_bf16(false, a, false, b, (short)0, c,
                                                   false, false);
}

__device__ __forceinline__ v8bf cvt8(float4 f0, float4 f1) {
    v8bf t;
    t[0] = (__bf16)f0.x; t[1] = (__bf16)f0.y; t[2] = (__bf16)f0.z; t[3] = (__bf16)f0.w;
    t[4] = (__bf16)f1.x; t[5] = (__bf16)f1.y; t[6] = (__bf16)f1.z; t[7] = (__bf16)f1.w;
    return t;
}

// ---------------------------------------------------------------------------
// Kernel 1: qkv = x @ Wqkv + bqkv   (M=4096, K=1024, N=3072)
// Block = 256 thr (8 waves), tile 128(M) x 128(N), K-step 32.
// Waves 4(M) x 2(N); each wave: 32M x 64N = 2 A-frags x 4 B-frags = 8 WMMA/step.
// Q/K/V routing is hoisted to ONE uniform scalar branch (128-wide N tiles never
// straddle a 1024-column segment; j*16+ln < 64 keeps `h` per-wave constant).
// ---------------------------------------------------------------------------
__global__ __launch_bounds__(256)
void qkv_gemm_kernel(const float* __restrict__ X, const float* __restrict__ W,
                     const float* __restrict__ bias,
                     __bf16* __restrict__ Qb, __bf16* __restrict__ Kb,
                     __bf16* __restrict__ Vtb)
{
    constexpr int K = CDIM;
    constexpr int N = 3 * CDIM;
    constexpr int AS = 40, BS = 40;                  // padded LDS strides (16B-aligned rows)
    __shared__ __align__(16) __bf16 ldsA[128 * AS];  // A tile  [128 M][32 K]
    __shared__ __align__(16) __bf16 ldsBt[128 * BS]; // B tile transposed [128 N][32 K]

    const int t = threadIdx.x;
    const int wave = t >> 5, lane = t & 31;
    const int half = lane >> 4, ln = lane & 15;
    const int mw = wave >> 1, nw = wave & 1;
    const int mblk = blockIdx.x * 128, nblk = blockIdx.y * 128;

    const int ar = t >> 1, ac = (t & 1) * 16;        // A staging (16 elems/thread)
    const int kr = t & 31, nc = (t >> 5) * 16;       // B staging: consecutive-bank stores

    v8f acc[2][4];
#pragma unroll
    for (int mi = 0; mi < 2; ++mi)
#pragma unroll
        for (int j = 0; j < 4; ++j)
#pragma unroll
            for (int r = 0; r < 8; ++r) acc[mi][j][r] = 0.0f;

    for (int kt = 0; kt < K; kt += 32) {
        if (kt + 32 < K) { // prefetch next K-tile (global_prefetch_b8)
            __builtin_prefetch(X + (size_t)(mblk + ar) * K + kt + 32 + ac, 0, 3);
            __builtin_prefetch(W + (size_t)(kt + 32 + kr) * N + nblk + nc, 0, 3);
        }
        { // stage A: fp32 -> bf16, row-major
            const float4* src =
                reinterpret_cast<const float4*>(X + (size_t)(mblk + ar) * K + kt + ac);
            *reinterpret_cast<v8bf*>(&ldsA[ar * AS + ac])     = cvt8(src[0], src[1]);
            *reinterpret_cast<v8bf*>(&ldsA[ar * AS + ac + 8]) = cvt8(src[2], src[3]);
        }
        { // stage B: fp32 -> bf16, transposed into [N][K]
            const float* src = W + (size_t)(kt + kr) * N + nblk + nc;
#pragma unroll
            for (int i = 0; i < 16; ++i) ldsBt[(nc + i) * BS + kr] = (__bf16)src[i];
        }
        __syncthreads();

        v16bf af[2];
#pragma unroll
        for (int mi = 0; mi < 2; ++mi) { // A frag: M = ln, K = e + half*16
            const __bf16* arow = &ldsA[(mw * 32 + mi * 16 + ln) * AS];
            af[mi] = pack16(*(const v8bf*)(arow + half * 16),
                            *(const v8bf*)(arow + half * 16 + 8));
        }
#pragma unroll
        for (int j = 0; j < 4; ++j) {
            // B frag: N = ln, K = e + half*16 (contiguous in transposed tile)
            const __bf16* brow = &ldsBt[(nw * 64 + j * 16 + ln) * BS + half * 16];
            v16bf bf = pack16(*(const v8bf*)(brow), *(const v8bf*)(brow + 8));
#pragma unroll
            for (int mi = 0; mi < 2; ++mi)
                acc[mi][j] = wmma_bf16(af[mi], bf, acc[mi][j]);
        }
        __syncthreads();
    }

    // Epilogue: one uniform branch, three straight-line store sequences.
    const int which = nblk >> 10;                    // block-uniform
    const int h = ((nblk + nw * 64) & 1023) >> 6;    // wave-uniform
    __bf16 bv[2][4][8];
#pragma unroll
    for (int j = 0; j < 4; ++j) {
        const float b = bias[nblk + nw * 64 + j * 16 + ln];
#pragma unroll
        for (int mi = 0; mi < 2; ++mi)
#pragma unroll
            for (int r = 0; r < 8; ++r) bv[mi][j][r] = (__bf16)(acc[mi][j][r] + b);
    }
    const int row0 = mblk + mw * 32 + half * 8;

    if (which == 0) {
        __bf16* base = Qb + ((size_t)h * LSEQ + row0) * DH + ln;
#pragma unroll
        for (int j = 0; j < 4; ++j)
#pragma unroll
            for (int mi = 0; mi < 2; ++mi)
#pragma unroll
                for (int r = 0; r < 8; ++r)
                    base[(size_t)(mi * 16 + r) * DH + j * 16] = bv[mi][j][r];
    } else if (which == 1) {
        __bf16* base = Kb + ((size_t)h * LSEQ + row0) * DH + ln;
#pragma unroll
        for (int j = 0; j < 4; ++j)
#pragma unroll
            for (int mi = 0; mi < 2; ++mi)
#pragma unroll
                for (int r = 0; r < 8; ++r)
                    base[(size_t)(mi * 16 + r) * DH + j * 16] = bv[mi][j][r];
    } else {
        __bf16* base = Vtb + ((size_t)h * DH + ln) * LSEQ + row0;
#pragma unroll
        for (int j = 0; j < 4; ++j)
#pragma unroll
            for (int mi = 0; mi < 2; ++mi)
#pragma unroll
                for (int r = 0; r < 8; ++r)
                    base[(size_t)(j * 16) * LSEQ + mi * 16 + r] = bv[mi][j][r];
    }
}

// ---------------------------------------------------------------------------
// Kernel 2: causal flash attention. One wave per (head, 16-query block).
// 32-key blocks: 4 WMMAs for S = Q K^T, online softmax, 4 WMMAs for O += P V.
// ---------------------------------------------------------------------------
__global__ __launch_bounds__(32)
void fa_kernel(const __bf16* __restrict__ Qb, const __bf16* __restrict__ Kb,
               const __bf16* __restrict__ Vtb, __bf16* __restrict__ AO)
{
    __shared__ __align__(16) __bf16 pTile[16 * 32]; // P in row-major for C->A relayout
    const int lane = threadIdx.x & 31;
    const int half = lane >> 4, ln = lane & 15;
    const int h = blockIdx.y;
    const int q0 = blockIdx.x * 16;

    // Q fragments (d 0..31 and 32..63), resident for whole kernel
    v16bf aq[2];
    {
        const __bf16* qrow = Qb + ((size_t)h * LSEQ + q0 + ln) * DH;
#pragma unroll
        for (int ks = 0; ks < 2; ++ks)
            aq[ks] = pack16(*(const v8bf*)(qrow + ks * 32 + half * 16),
                            *(const v8bf*)(qrow + ks * 32 + half * 16 + 8));
    }

    float m[8], l[8];
    v8f o[4];
#pragma unroll
    for (int r = 0; r < 8; ++r) { m[r] = -3.0e38f; l[r] = 0.0f; }
#pragma unroll
    for (int j = 0; j < 4; ++j)
#pragma unroll
        for (int r = 0; r < 8; ++r) o[j][r] = 0.0f;

    const int nkb = (q0 + 47) >> 5; // 32-key blocks covering keys <= q0+15
    for (int kb = 0; kb < nkb; ++kb) {
        const int k0 = kb * 32;

        if (kb + 1 < nkb) { // prefetch next key/value block rows
            __builtin_prefetch(Kb + ((size_t)h * LSEQ + k0 + 32 + lane) * DH, 0, 3);
            __builtin_prefetch(Vtb + ((size_t)h * DH + 2 * lane) * LSEQ + k0 + 32, 0, 3);
        }

        v8f s[2];
#pragma unroll
        for (int ns = 0; ns < 2; ++ns)
#pragma unroll
            for (int r = 0; r < 8; ++r) s[ns][r] = 0.0f;

        // S = Q K^T : B fragment N=key (ln), depth=d (e + half*16), contiguous in K rows
#pragma unroll
        for (int ns = 0; ns < 2; ++ns) {
            const __bf16* krow =
                Kb + ((size_t)h * LSEQ + k0 + ns * 16 + ln) * DH + half * 16;
#pragma unroll
            for (int ks = 0; ks < 2; ++ks) {
                v16bf bk = pack16(*(const v8bf*)(krow + ks * 32),
                                  *(const v8bf*)(krow + ks * 32 + 8));
                s[ns] = wmma_bf16(aq[ks], bk, s[ns]);
            }
        }

        // scale 1/sqrt(64) + causal mask in C layout (row = r+half*8, col = ln)
        const bool needMask = (k0 + 31) > q0;
#pragma unroll
        for (int ns = 0; ns < 2; ++ns) {
            const int key = k0 + ns * 16 + ln;
#pragma unroll
            for (int r = 0; r < 8; ++r) {
                float v = s[ns][r] * 0.125f;
                if (needMask && key > (q0 + r + half * 8)) v = -3.0e38f;
                s[ns][r] = v;
            }
        }

        // online softmax: row max across the 16-lane half
        float alpha[8];
#pragma unroll
        for (int r = 0; r < 8; ++r) {
            float v = fmaxf(s[0][r], s[1][r]);
            v = fmaxf(v, __shfl_xor(v, 1));
            v = fmaxf(v, __shfl_xor(v, 2));
            v = fmaxf(v, __shfl_xor(v, 4));
            v = fmaxf(v, __shfl_xor(v, 8));
            const float mn = fmaxf(m[r], v);
            alpha[r] = __expf(m[r] - mn);
            m[r] = mn;
        }
#pragma unroll
        for (int ns = 0; ns < 2; ++ns)
#pragma unroll
            for (int r = 0; r < 8; ++r) {
                const float p = __expf(s[ns][r] - m[r]);
                s[ns][r] = p;
                pTile[(r + half * 8) * 32 + ns * 16 + ln] = (__bf16)p;
            }
#pragma unroll
        for (int r = 0; r < 8; ++r) {
            float v = s[0][r] + s[1][r];
            v += __shfl_xor(v, 1);
            v += __shfl_xor(v, 2);
            v += __shfl_xor(v, 4);
            v += __shfl_xor(v, 8);
            l[r] = l[r] * alpha[r] + v;
        }
#pragma unroll
        for (int j = 0; j < 4; ++j)
#pragma unroll
            for (int r = 0; r < 8; ++r) o[j][r] *= alpha[r];

        __syncthreads(); // single-wave WG: DS in-order, barrier is cheap/NOP

        // P as A fragment from LDS (row = ln, K = e + half*16)
        v16bf pa;
        {
            const __bf16* prow = &pTile[ln * 32];
            pa = pack16(*(const v8bf*)(prow + half * 16),
                        *(const v8bf*)(prow + half * 16 + 8));
        }
        // O += P V : V^T rows give contiguous per-lane B fragments (depth = key)
#pragma unroll
        for (int j = 0; j < 4; ++j) {
            const __bf16* vrow =
                Vtb + ((size_t)h * DH + j * 16 + ln) * LSEQ + k0 + half * 16;
            v16bf bv = pack16(*(const v8bf*)(vrow), *(const v8bf*)(vrow + 8));
            o[j] = wmma_bf16(pa, bv, o[j]);
        }
        __syncthreads();
    }

#pragma unroll
    for (int r = 0; r < 8; ++r) l[r] = 1.0f / l[r];
#pragma unroll
    for (int j = 0; j < 4; ++j) {
        const int col = h * DH + j * 16 + ln;
#pragma unroll
        for (int r = 0; r < 8; ++r) {
            const int row = q0 + r + half * 8;
            AO[(size_t)row * CDIM + col] = (__bf16)(o[j][r] * l[r]);
        }
    }
}

// ---------------------------------------------------------------------------
// Kernel 3: out = attn_out(bf16) @ Wout + bout   (M=4096, K=1024, N=1024), fp32 out
// Same 128x128 block tile structure as kernel 1.
// ---------------------------------------------------------------------------
__global__ __launch_bounds__(256)
void out_gemm_kernel(const __bf16* __restrict__ A, const float* __restrict__ W,
                     const float* __restrict__ bias, float* __restrict__ Out)
{
    constexpr int K = CDIM;
    constexpr int N = CDIM;
    constexpr int AS = 40, BS = 40;
    __shared__ __align__(16) __bf16 ldsA[128 * AS];
    __shared__ __align__(16) __bf16 ldsBt[128 * BS];

    const int t = threadIdx.x;
    const int wave = t >> 5, lane = t & 31;
    const int half = lane >> 4, ln = lane & 15;
    const int mw = wave >> 1, nw = wave & 1;
    const int mblk = blockIdx.x * 128, nblk = blockIdx.y * 128;

    const int ar = t >> 1, ac = (t & 1) * 16;
    const int kr = t & 31, nc = (t >> 5) * 16;

    v8f acc[2][4];
#pragma unroll
    for (int mi = 0; mi < 2; ++mi)
#pragma unroll
        for (int j = 0; j < 4; ++j)
#pragma unroll
            for (int r = 0; r < 8; ++r) acc[mi][j][r] = 0.0f;

    for (int kt = 0; kt < K; kt += 32) {
        if (kt + 32 < K) {
            __builtin_prefetch(A + (size_t)(mblk + ar) * K + kt + 32 + ac, 0, 3);
            __builtin_prefetch(W + (size_t)(kt + 32 + kr) * N + nblk + nc, 0, 3);
        }
        { // stage A: already bf16, two 16B loads per thread
            const v8bf* src =
                reinterpret_cast<const v8bf*>(A + (size_t)(mblk + ar) * K + kt + ac);
            *reinterpret_cast<v8bf*>(&ldsA[ar * AS + ac])     = src[0];
            *reinterpret_cast<v8bf*>(&ldsA[ar * AS + ac + 8]) = src[1];
        }
        { // stage B: fp32 -> bf16, transposed
            const float* src = W + (size_t)(kt + kr) * N + nblk + nc;
#pragma unroll
            for (int i = 0; i < 16; ++i) ldsBt[(nc + i) * BS + kr] = (__bf16)src[i];
        }
        __syncthreads();

        v16bf af[2];
#pragma unroll
        for (int mi = 0; mi < 2; ++mi) {
            const __bf16* arow = &ldsA[(mw * 32 + mi * 16 + ln) * AS];
            af[mi] = pack16(*(const v8bf*)(arow + half * 16),
                            *(const v8bf*)(arow + half * 16 + 8));
        }
#pragma unroll
        for (int j = 0; j < 4; ++j) {
            const __bf16* brow = &ldsBt[(nw * 64 + j * 16 + ln) * BS + half * 16];
            v16bf bf = pack16(*(const v8bf*)(brow), *(const v8bf*)(brow + 8));
#pragma unroll
            for (int mi = 0; mi < 2; ++mi)
                acc[mi][j] = wmma_bf16(af[mi], bf, acc[mi][j]);
        }
        __syncthreads();
    }

#pragma unroll
    for (int j = 0; j < 4; ++j) {
        const int c = nblk + nw * 64 + j * 16 + ln;
        const float b = bias[c];
#pragma unroll
        for (int mi = 0; mi < 2; ++mi) {
#pragma unroll
            for (int r = 0; r < 8; ++r) {
                const int row = mblk + mw * 32 + mi * 16 + r + half * 8;
                Out[(size_t)row * N + c] = acc[mi][j][r] + b;
            }
        }
    }
}

// ---------------------------------------------------------------------------
extern "C" void kernel_launch(void* const* d_in, const int* in_sizes, int n_in,
                              void* d_out, int out_size, void* d_ws, size_t ws_size,
                              hipStream_t stream)
{
    const float* x    = (const float*)d_in[0]; // [4096, 1024]
    const float* Wqkv = (const float*)d_in[1]; // [1024, 3072]
    const float* bqkv = (const float*)d_in[2]; // [3072]
    const float* Wout = (const float*)d_in[3]; // [1024, 1024]
    const float* bout = (const float*)d_in[4]; // [1024]
    float* out = (float*)d_out;                // [4096, 1024] fp32

    // Workspace: Q, K (head-major [H][L][D]) + V^T ([H][D][L]) + attn-out, all bf16.
    char* ws = (char*)d_ws;
    const size_t seg = (size_t)HN * LSEQ * DH * sizeof(__bf16); // 8 MB each
    __bf16* Qb  = (__bf16*)(ws);
    __bf16* Kb  = (__bf16*)(ws + seg);
    __bf16* Vtb = (__bf16*)(ws + 2 * seg);
    __bf16* AO  = (__bf16*)(ws + 3 * seg);

    qkv_gemm_kernel<<<dim3(LSEQ / 128, (3 * CDIM) / 128), 256, 0, stream>>>(
        x, Wqkv, bqkv, Qb, Kb, Vtb);
    fa_kernel<<<dim3(LSEQ / 16, HN), 32, 0, stream>>>(Qb, Kb, Vtb, AO);
    out_gemm_kernel<<<dim3(LSEQ / 128, CDIM / 128), 256, 0, stream>>>(
        AO, Wout, bout, out);
}
